// DiffusionUNet_64_36507222016351
// MI455X (gfx1250) — matrix operand order repaired and checked by reading, not compiled, not run on hardware
//
#include <hip/hip_runtime.h>
#include <hip/hip_bf16.h>
#include <stdint.h>

// ---------------- problem constants ----------------
#define B_    32
#define CIN_  256
#define COUT_ 256
#define H_    32
#define W_    32
#define E_    4
#define TDIM_ 256
#define HID_  64

#if defined(__has_builtin) && __has_builtin(__builtin_amdgcn_sched_barrier)
#define SCHED_FENCE() __builtin_amdgcn_sched_barrier(0)
#else
#define SCHED_FENCE() ((void)0)
#endif

typedef __attribute__((ext_vector_type(16))) __bf16 bf16x16;
typedef __attribute__((ext_vector_type(8)))  float  f32x8;

union Frag {
    bf16x16 v;
    uint4   q[2];
};
static_assert(sizeof(Frag) == 32, "frag size");

__device__ __forceinline__ uint16_t f2bf(float f) {
    uint32_t u = __float_as_uint(f);
    u += 0x7FFFu + ((u >> 16) & 1u);   // round-to-nearest-even
    return (uint16_t)(u >> 16);
}

// ---------------- 1) global average pool: x(B,CIN,32,32) -> pooled(B,CIN) ----
__global__ void pool_kernel(const float* __restrict__ x, float* __restrict__ pooled) {
    int wave = threadIdx.x >> 5, lane = threadIdx.x & 31;
    int pair = blockIdx.x * 8 + wave;                 // pair = b*CIN + ci
    const float* p = x + (size_t)pair * (H_ * W_);
    float s = 0.f;
    #pragma unroll
    for (int i = 0; i < 32; ++i) s += p[lane + 32 * i];
    #pragma unroll
    for (int off = 16; off > 0; off >>= 1) s += __shfl_xor(s, off, 32);
    if (lane == 0) pooled[pair] = s * (1.0f / (H_ * W_));
}

// ---------------- 2) router MLP + softmax -> rw(B,E) ------------------------
__global__ void router_kernel(const float* __restrict__ te, const float* __restrict__ pooled,
                              const float* __restrict__ Wq, const float* __restrict__ bq,
                              const float* __restrict__ Wk, const float* __restrict__ bk,
                              const float* __restrict__ Wv, const float* __restrict__ bv,
                              const float* __restrict__ Wm1, const float* __restrict__ bm1,
                              const float* __restrict__ Wm2, const float* __restrict__ bm2,
                              const float* __restrict__ Wc, const float* __restrict__ bc,
                              float* __restrict__ rw) {
    __shared__ float xa[B_ * HID_];
    __shared__ float h1[B_ * HID_];
    __shared__ float xm[B_ * HID_];
    __shared__ float lg[B_ * E_];
    int tid = threadIdx.x;

    // q,k,v,attn,xa per (b,h)
    for (int i = 0; i < 8; ++i) {
        int pair = tid + i * 256;
        int b = pair >> 6, h = pair & 63;
        const float* teb = te + b * TDIM_;
        const float* pb  = pooled + b * CIN_;
        const float* wq = Wq + h * TDIM_;
        const float* wk = Wk + h * CIN_;
        const float* wv = Wv + h * CIN_;
        float q = bq[h], k = bk[h], v = bv[h];
        for (int t = 0; t < TDIM_; ++t) q += teb[t] * wq[t];
        for (int t = 0; t < CIN_; ++t) { k += pb[t] * wk[t]; v += pb[t] * wv[t]; }
        float attn = 1.0f / (1.0f + __expf(-(q * k)));
        xa[pair] = v * attn;
    }
    __syncthreads();
    // h1 = silu(xa @ Wm1^T + bm1)
    for (int i = 0; i < 8; ++i) {
        int pair = tid + i * 256;
        int b = pair >> 6, h = pair & 63;
        const float* w  = Wm1 + h * HID_;
        const float* xr = xa + b * HID_;
        float a = bm1[h];
        for (int t = 0; t < HID_; ++t) a += xr[t] * w[t];
        h1[pair] = a / (1.0f + __expf(-a));
    }
    __syncthreads();
    // xm = xa + h1 @ Wm2^T + bm2
    for (int i = 0; i < 8; ++i) {
        int pair = tid + i * 256;
        int b = pair >> 6, h = pair & 63;
        const float* w  = Wm2 + h * HID_;
        const float* hr = h1 + b * HID_;
        float a = bm2[h];
        for (int t = 0; t < HID_; ++t) a += hr[t] * w[t];
        xm[pair] = xa[pair] + a;
    }
    __syncthreads();
    // logits = xm @ Wc^T + bc
    if (tid < B_ * E_) {
        int b = tid >> 2, e = tid & 3;
        const float* w  = Wc + e * HID_;
        const float* xr = xm + b * HID_;
        float a = bc[e];
        for (int t = 0; t < HID_; ++t) a += xr[t] * w[t];
        lg[tid] = a;
    }
    __syncthreads();
    // softmax over E=4
    if (tid < B_) {
        float m = lg[tid * 4];
        #pragma unroll
        for (int e = 1; e < 4; ++e) m = fmaxf(m, lg[tid * 4 + e]);
        float ex[4], s = 0.f;
        #pragma unroll
        for (int e = 0; e < 4; ++e) { ex[e] = __expf(lg[tid * 4 + e] - m); s += ex[e]; }
        #pragma unroll
        for (int e = 0; e < 4; ++e) rw[tid * 4 + e] = ex[e] / s;
    }
}

// ---------------- 3) expert mix -> bf16 weights in WMMA-A friendly layout ---
// wA[b][co][tap][ci] = bf16( sum_e rw[b,e] * weight[e][co][ci][kh][kw] ), tap=kh*3+kw
__global__ void mix_kernel(const float* __restrict__ weight, const float* __restrict__ rw,
                           uint16_t* __restrict__ wA) {
    size_t idx = (size_t)blockIdx.x * 256 + threadIdx.x;   // [b][co][tap][ci]
    int ci  = (int)(idx & 255);
    int tap = (int)((idx >> 8) % 9);
    int co  = (int)((idx / 2304) & 255);
    int b   = (int)(idx / 589824);
    float r0 = rw[b * 4 + 0], r1 = rw[b * 4 + 1], r2 = rw[b * 4 + 2], r3 = rw[b * 4 + 3];
    size_t wi = ((size_t)co * CIN_ + ci) * 9 + tap;        // index within one expert
    const size_t es = (size_t)COUT_ * CIN_ * 9;
    float v = r0 * weight[wi] + r1 * weight[wi + es] + r2 * weight[wi + 2 * es] + r3 * weight[wi + 3 * es];
    wA[idx] = f2bf(v);
}

// ---------------- 4) per-sample implicit-GEMM conv via bf16 WMMA ------------
// Block tile: M=128 (co), N=128 (pixels = 4 image rows). 8 waves: 2(M) x 4(N rows).
// K loop: 8 chunks of 32 ci x 9 taps; x tile staged in LDS as [6][34][32ch] bf16.
// Tap loop is software-pipelined; sched_barrier pins the prefetch loads for tap
// t+1 ahead of the WMMA block of tap t so the waits resolve under compute.
#define MT 4
#define NT 2

__global__ __launch_bounds__(256)
void conv_kernel(const float* __restrict__ x, const uint16_t* __restrict__ wA,
                 float* __restrict__ out) {
    __shared__ uint16_t xs[6 * 34 * 32];     // [r][c][ch], ch fastest (12.75 KB)

    int tid   = threadIdx.x;
    int lane  = tid & 31;
    int wave  = tid >> 5;
    int lane16 = lane & 15;
    int half   = lane >> 4;
    int wave_m = wave >> 2;     // 0..1
    int wave_n = wave & 3;      // 0..3  (== image row within N tile)

    int bx = blockIdx.x;
    int nb = bx & 7;            // 8 N-blocks of 128 pixels (4 rows)
    int mb = (bx >> 3) & 1;     // 2 M-blocks of 128 co
    int b  = bx >> 4;           // sample
    int y0  = nb * 4;
    int co0 = mb * 128;

    f32x8 acc[MT][NT];
    #pragma unroll
    for (int i = 0; i < MT; ++i)
        #pragma unroll
        for (int j = 0; j < NT; ++j)
            #pragma unroll
            for (int v = 0; v < 8; ++v) acc[i][j][v] = 0.0f;

    // per-lane A base pointers (everything but tap/ci0 folded in)
    const uint16_t* baseA[MT];
    #pragma unroll
    for (int mt = 0; mt < MT; ++mt) {
        int co = co0 + wave_m * 64 + mt * 16 + lane16;
        baseA[mt] = wA + (((size_t)(b * COUT_ + co) * 9) << 8) + half * 8;
    }

    Frag afr[2][MT];
    Frag bfr[2][NT];

    auto loadA = [&](Frag* af, int tap, int ci0) {
        #pragma unroll
        for (int mt = 0; mt < MT; ++mt) {
            const uint16_t* p = baseA[mt] + (tap << 8) + ci0;
            af[mt].q[0] = *(const uint4*)(p);
            af[mt].q[1] = *(const uint4*)(p + 16);
        }
    };
    auto loadB = [&](Frag* bf, int tap) {
        int dy = tap / 3, dx = tap - dy * 3;
        int r = wave_n + dy;
        #pragma unroll
        for (int nt = 0; nt < NT; ++nt) {
            int c = nt * 16 + lane16 + dx;
            const uint16_t* cell = &xs[(r * 34 + c) * 32 + half * 8];
            bf[nt].q[0] = *(const uint4*)(cell);
            bf[nt].q[1] = *(const uint4*)(cell + 16);
        }
    };

    for (int ci0 = 0; ci0 < CIN_; ci0 += 32) {
        __syncthreads();
        // stage x tile (rows y0-1..y0+4, cols -1..32, 32 channels) -> bf16 LDS
        for (int e = tid; e < 6 * 34 * 32; e += 256) {
            int c  = e % 34;
            int r  = (e / 34) % 6;
            int ch = e / 204;
            int gy = y0 + r - 1, gx = c - 1;
            float v = 0.0f;
            if ((unsigned)gy < 32u && (unsigned)gx < 32u)
                v = x[(((size_t)b * CIN_ + ci0 + ch) << 10) + (gy << 5) + gx];
            xs[(r * 34 + c) * 32 + ch] = f2bf(v);
        }
        __syncthreads();

        // pipeline prologue: fragments for tap 0
        loadA(afr[0], 0, ci0);
        loadB(bfr[0], 0);

        #pragma unroll
        for (int tap = 0; tap < 9; ++tap) {
            int p = tap & 1, q = p ^ 1;
            if (tap + 1 < 9) {              // prefetch next tap under the WMMAs
                loadA(afr[q], tap + 1, ci0);
                loadB(bfr[q], tap + 1);
            }
            SCHED_FENCE();                  // loads for t+1 stay issued before WMMAs of t
            #pragma unroll
            for (int mt = 0; mt < MT; ++mt)
                #pragma unroll
                for (int nt = 0; nt < NT; ++nt) {
                    acc[mt][nt] = __builtin_amdgcn_wmma_f32_16x16x32_bf16(
                        false, afr[p][mt].v, false, bfr[p][nt].v,
                        (short)0, acc[mt][nt], false, false);
                }
            SCHED_FENCE();                  // keep iteration bodies from collapsing
        }
    }

    // store: C/D layout -> lanes 0-15: M=v, N=lane16; lanes 16-31: M=v+8
    int p0 = nb * 128;
    #pragma unroll
    for (int mt = 0; mt < MT; ++mt)
        #pragma unroll
        for (int nt = 0; nt < NT; ++nt) {
            int n_local = wave_n * 32 + nt * 16 + lane16;
            #pragma unroll
            for (int v = 0; v < 8; ++v) {
                int co = co0 + wave_m * 64 + mt * 16 + v + half * 8;
                out[(((size_t)b * COUT_ + co) << 10) + p0 + n_local] = acc[mt][nt][v];
            }
        }
}

// ---------------- launcher ---------------------------------------------------
extern "C" void kernel_launch(void* const* d_in, const int* in_sizes, int n_in,
                              void* d_out, int out_size, void* d_ws, size_t ws_size,
                              hipStream_t stream) {
    const float* x    = (const float*)d_in[0];
    const float* te   = (const float*)d_in[1];
    const float* w    = (const float*)d_in[2];
    const float* Wq   = (const float*)d_in[3];
    const float* bq   = (const float*)d_in[4];
    const float* Wk   = (const float*)d_in[5];
    const float* bk   = (const float*)d_in[6];
    const float* Wv   = (const float*)d_in[7];
    const float* bv   = (const float*)d_in[8];
    const float* Wm1  = (const float*)d_in[9];
    const float* bm1  = (const float*)d_in[10];
    const float* Wm2  = (const float*)d_in[11];
    const float* bm2  = (const float*)d_in[12];
    const float* Wc   = (const float*)d_in[13];
    const float* bc   = (const float*)d_in[14];
    float* out = (float*)d_out;

    // workspace layout
    float*    rw     = (float*)d_ws;                                // 512 B
    float*    pooled = (float*)((char*)d_ws + 4096);                // 32 KB
    uint16_t* wA     = (uint16_t*)((char*)d_ws + 65536);            // 36 MB bf16

    pool_kernel<<<dim3((B_ * CIN_) / 8), dim3(256), 0, stream>>>(x, pooled);
    router_kernel<<<dim3(1), dim3(256), 0, stream>>>(te, pooled, Wq, bq, Wk, bk, Wv, bv,
                                                     Wm1, bm1, Wm2, bm2, Wc, bc, rw);
    mix_kernel<<<dim3((B_ * COUT_ * 9 * CIN_) / 256), dim3(256), 0, stream>>>(w, rw, wA);
    conv_kernel<<<dim3(B_ * 2 * 8), dim3(256), 0, stream>>>(x, wA, out);
}
